// GraphAttentionEncoder_4294967296541
// MI455X (gfx1250) — matrix operand order repaired and checked by reading, not compiled, not run on hardware
//
#include <hip/hip_runtime.h>
#include <hip/hip_bf16.h>

#define NN 20000
#define DD 256
#define KK 16
#define RD 64

typedef __attribute__((ext_vector_type(16))) __bf16 v16bf;
typedef __attribute__((ext_vector_type(8)))  __bf16 v8bf;
typedef __attribute__((ext_vector_type(8)))  float  v8f;

// ---- fragment loaders (CDNA5 WMMA 16-bit layouts, CDNA5 ISA) ----
// A matrix 16x32: lanes 0-15 -> K {0..7,16..23}, lanes 16-31 -> K {8..15,24..31}
static __device__ __forceinline__ v16bf load_a_frag(const __bf16* row, int kb, int half) {
    int k0 = kb * 32 + half * 8;
    v8bf lo = *(const v8bf*)(row + k0);
    v8bf hi = *(const v8bf*)(row + k0 + 16);
    v16bf r;
#pragma unroll
    for (int j = 0; j < 8; ++j) { r[j] = lo[j]; r[j + 8] = hi[j]; }
    return r;
}
// B matrix 32x16: lanes 0-15 -> K {0..7,16..23}, lanes 16-31 -> K {8..15,24..31}
static __device__ __forceinline__ v16bf load_b_frag(const __bf16* row, int kb, int half) {
    int k0 = kb * 32 + half * 8;
    v8bf lo = *(const v8bf*)(row + k0);
    v8bf hi = *(const v8bf*)(row + k0 + 16);
    v16bf r;
#pragma unroll
    for (int j = 0; j < 8; ++j) { r[j] = lo[j]; r[j + 8] = hi[j]; }
    return r;
}

static __device__ __forceinline__ void ld8(const float* p, float* v) {
    float4 a = *(const float4*)p;
    float4 b = *(const float4*)(p + 4);
    v[0] = a.x; v[1] = a.y; v[2] = a.z; v[3] = a.w;
    v[4] = b.x; v[5] = b.y; v[6] = b.z; v[7] = b.w;
}

// ---------------- Kernel 1: row LayerNorm, one wave32 per row ----------------
__global__ void ln_kernel(const float* __restrict__ x, const float* __restrict__ g,
                          const float* __restrict__ b, float* __restrict__ t,
                          __bf16* __restrict__ tb) {
    int wid = threadIdx.x >> 5, lane = threadIdx.x & 31;
    int row = blockIdx.x * 8 + wid;              // 20000/8 = 2500 exact
    const float* xr = x + (size_t)row * DD + lane * 8;
    float4 a0 = *(const float4*)xr;
    float4 a1 = *(const float4*)(xr + 4);
    float s = a0.x + a0.y + a0.z + a0.w + a1.x + a1.y + a1.z + a1.w;
    float q = a0.x*a0.x + a0.y*a0.y + a0.z*a0.z + a0.w*a0.w
            + a1.x*a1.x + a1.y*a1.y + a1.z*a1.z + a1.w*a1.w;
#pragma unroll
    for (int off = 16; off > 0; off >>= 1) {
        s += __shfl_xor(s, off, 32);
        q += __shfl_xor(q, off, 32);
    }
    float mean = s * (1.0f / DD);
    float var  = q * (1.0f / DD) - mean * mean;
    float rstd = rsqrtf(var + 1e-5f);
    const float* gp = g + lane * 8;
    const float* bp = b + lane * 8;
    float4 g0 = *(const float4*)gp, g1 = *(const float4*)(gp + 4);
    float4 b0 = *(const float4*)bp, b1 = *(const float4*)(bp + 4);
    float o[8];
    o[0] = (a0.x - mean) * rstd * g0.x + b0.x;
    o[1] = (a0.y - mean) * rstd * g0.y + b0.y;
    o[2] = (a0.z - mean) * rstd * g0.z + b0.z;
    o[3] = (a0.w - mean) * rstd * g0.w + b0.w;
    o[4] = (a1.x - mean) * rstd * g1.x + b1.x;
    o[5] = (a1.y - mean) * rstd * g1.y + b1.y;
    o[6] = (a1.z - mean) * rstd * g1.z + b1.z;
    o[7] = (a1.w - mean) * rstd * g1.w + b1.w;
    float* tr = t + (size_t)row * DD + lane * 8;
    *(float4*)tr       = make_float4(o[0], o[1], o[2], o[3]);
    *(float4*)(tr + 4) = make_float4(o[4], o[5], o[6], o[7]);
    __bf16* tbr = tb + (size_t)row * DD + lane * 8;
#pragma unroll
    for (int j = 0; j < 8; ++j) tbr[j] = (__bf16)o[j];
}

// ---------------- f32 -> bf16 weight conversion ----------------
__global__ void cvt_kernel(const float* __restrict__ in, __bf16* __restrict__ o, int n) {
    int idx = blockIdx.x * 256 + threadIdx.x;
    if (idx < n) o[idx] = (__bf16)in[idx];
}

// ---- Kernel 2: dual GEMM  s = t@Wselfᵀ+b,  ns = t@Wneiᵀ+b  (A-frag reuse) ----
__global__ void gemm2_kernel(const __bf16* __restrict__ tb,
                             const __bf16* __restrict__ wsb, const __bf16* __restrict__ wnb,
                             const float* __restrict__ bs, const float* __restrict__ bn,
                             float* __restrict__ s_sc, float* __restrict__ ns) {
    int wid = threadIdx.x >> 5, lane = threadIdx.x & 31;
    int it = blockIdx.x;                 // row tile, 1250
    int jt = blockIdx.y * 4 + wid;       // col tile, 0..15
    int m = lane & 15, half = lane >> 4;
    const __bf16* arow  = tb  + (size_t)(it * 16 + m) * DD;
    const __bf16* bsrow = wsb + (size_t)(jt * 16 + m) * DD;   // B[k][n] = W[n][k]
    const __bf16* bnrow = wnb + (size_t)(jt * 16 + m) * DD;
    v8f accs = {};
    v8f accn = {};
#pragma unroll
    for (int kb = 0; kb < 8; ++kb) {     // K = 256 / 32
        v16bf a  = load_a_frag(arow, kb, half);
        v16bf b1 = load_b_frag(bsrow, kb, half);
        v16bf b2 = load_b_frag(bnrow, kb, half);
        accs = __builtin_amdgcn_wmma_f32_16x16x32_bf16(false, a, false, b1, (short)0, accs, false, false);
        accn = __builtin_amdgcn_wmma_f32_16x16x32_bf16(false, a, false, b2, (short)0, accn, false, false);
    }
    int col = jt * 16 + m;
    float vbs = bs[col], vbn = bn[col];
#pragma unroll
    for (int r = 0; r < 8; ++r) {        // C/D layout: VGPR r -> M = r + 8*half
        int row = it * 16 + r + half * 8;
        s_sc[(size_t)row * DD + col] = accs[r] + vbs;
        ns  [(size_t)row * DD + col] = accn[r] + vbn;
    }
}

// ---- Kernel 3: softmax attention + context + fused 256->64 WMMA reduce ----
__global__ void attn_kernel(const float* __restrict__ s_sc, const float* __restrict__ ns,
                            const float* __restrict__ t, const float* __restrict__ coords,
                            const int* __restrict__ colidx, const float* __restrict__ beta_mix,
                            const __bf16* __restrict__ wrb, const float* __restrict__ brd,
                            float* __restrict__ out) {
    __shared__ int   jn[16][16];
    __shared__ float wn[16][16];
    __shared__ __align__(16) __bf16 ctx[16][264];   // +8 bf16 pad per row vs LDS banks

    int tid = threadIdx.x;
    int base = blockIdx.x * 16;
    {   // distance weights + neighbor indices: thread = (node, k)
        int nl = tid >> 4, k = tid & 15;
        int i = base + nl;
        int j = colidx[(size_t)i * KK + k];
        jn[nl][k] = j;
        float dx = coords[2 * i]     - coords[2 * j];
        float dy = coords[2 * i + 1] - coords[2 * j + 1];
        float dist = sqrtf(dx * dx + dy * dy);
        wn[nl][k] = expf(-2.0f * dist / (50.0f + 1e-8f));
    }
    __syncthreads();

    float bm = beta_mix[0];
    float om = 1.0f - bm;
    int wid = tid >> 5, lane = tid & 31;
    int d0 = lane * 8;                   // each lane owns 8 contiguous dims

    for (int nn = 0; nn < 2; ++nn) {     // 8 waves x 2 nodes = 16 nodes
        int nl = wid * 2 + nn;
        int i = base + nl;
        float sv[8], mx[8], den[8], es[8], cv[8];
        ld8(s_sc + (size_t)i * DD + d0, sv);
#pragma unroll
        for (int d = 0; d < 8; ++d) mx[d] = sv[d];
        // pass 1: running max over the 17-entry softmax axis (L2-resident gathers)
        for (int k = 0; k < KK; ++k) {
            int j = jn[nl][k]; float w = wn[nl][k];
            float v[8]; ld8(ns + (size_t)j * DD + d0, v);
#pragma unroll
            for (int d = 0; d < 8; ++d) mx[d] = fmaxf(mx[d], v[d] * w);
        }
        // pass 2: denominator
#pragma unroll
        for (int d = 0; d < 8; ++d) { es[d] = expf(sv[d] - mx[d]); den[d] = es[d]; }
        for (int k = 0; k < KK; ++k) {
            int j = jn[nl][k]; float w = wn[nl][k];
            float v[8]; ld8(ns + (size_t)j * DD + d0, v);
#pragma unroll
            for (int d = 0; d < 8; ++d) den[d] += expf(v[d] * w - mx[d]);
        }
        // pass 3: thresholded weights + context accumulation
        float ti[8]; ld8(t + (size_t)i * DD + d0, ti);
#pragma unroll
        for (int d = 0; d < 8; ++d) {
            float inv = 1.0f / den[d];
            den[d] = inv;
            float c0 = es[d] * inv;
            if (c0 < 0.01f) c0 = 0.0f;
            cv[d] = bm * c0 * ti[d];
        }
        for (int k = 0; k < KK; ++k) {
            int j = jn[nl][k]; float w = wn[nl][k];
            float v[8];  ld8(ns + (size_t)j * DD + d0, v);
            float tj[8]; ld8(t  + (size_t)j * DD + d0, tj);
#pragma unroll
            for (int d = 0; d < 8; ++d) {
                float e = expf(v[d] * w - mx[d]) * den[d];
                if (e < 0.01f) e = 0.0f;
                cv[d] += om * e * tj[d];
            }
        }
#pragma unroll
        for (int d = 0; d < 8; ++d) ctx[nl][d0 + d] = (__bf16)cv[d];
    }
    __syncthreads();

    // fused reduce GEMM: (16x256 ctx) x (256x64 W_redᵀ), waves 0..3 = 4 col tiles
    if (wid < 4) {
        int m = lane & 15, half = lane >> 4;
        const __bf16* arow = &ctx[m][0];
        const __bf16* brow = wrb + (size_t)(wid * 16 + m) * DD;
        v8f acc = {};
#pragma unroll
        for (int kb = 0; kb < 8; ++kb) {
            v16bf a = load_a_frag(arow, kb, half);
            v16bf b = load_b_frag(brow, kb, half);
            acc = __builtin_amdgcn_wmma_f32_16x16x32_bf16(false, a, false, b, (short)0, acc, false, false);
        }
        int colo = wid * 16 + m;
        float bb = brd[colo];
#pragma unroll
        for (int r = 0; r < 8; ++r) {
            int row = base + r + half * 8;
            float v = acc[r] + bb;
            v = v > 0.0f ? v : 0.01f * v;    // leaky_relu
            v = v > 0.0f ? v : 0.01f * v;    // applied twice in reference
            out[(size_t)row * RD + colo] = v;
        }
    }
}

extern "C" void kernel_launch(void* const* d_in, const int* in_sizes, int n_in,
                              void* d_out, int out_size, void* d_ws, size_t ws_size,
                              hipStream_t stream) {
    const float* x        = (const float*)d_in[0];
    const float* coords   = (const float*)d_in[1];
    const float* gamma    = (const float*)d_in[2];
    const float* beta_ln  = (const float*)d_in[3];
    const float* W_self   = (const float*)d_in[4];
    const float* b_self   = (const float*)d_in[5];
    const float* W_nei    = (const float*)d_in[6];
    const float* b_nei    = (const float*)d_in[7];
    const float* W_red    = (const float*)d_in[8];
    const float* b_red    = (const float*)d_in[9];
    const float* beta_mix = (const float*)d_in[10];
    const int*   edge_idx = (const int*)d_in[11];
    const int*   col      = edge_idx + (size_t)NN * KK;   // second row of edge_index
    float* out = (float*)d_out;

    const size_t NDf = (size_t)NN * DD;                   // 5,120,000
    float*  t        = (float*)d_ws;
    float*  s_sc     = t + NDf;
    float*  ns       = s_sc + NDf;
    __bf16* t_bf     = (__bf16*)(ns + NDf);
    __bf16* wself_bf = t_bf + NDf;
    __bf16* wnei_bf  = wself_bf + DD * DD;
    __bf16* wred_bf  = wnei_bf + DD * DD;                 // total ~72 MB

    ln_kernel<<<NN / 8, 256, 0, stream>>>(x, gamma, beta_ln, t, t_bf);
    cvt_kernel<<<(DD * DD) / 256, 256, 0, stream>>>(W_self, wself_bf, DD * DD);
    cvt_kernel<<<(DD * DD) / 256, 256, 0, stream>>>(W_nei,  wnei_bf,  DD * DD);
    cvt_kernel<<<(RD * DD) / 256, 256, 0, stream>>>(W_red,  wred_bf,  RD * DD);
    gemm2_kernel<<<dim3(NN / 16, 4), 128, 0, stream>>>(t_bf, wself_bf, wnei_bf,
                                                       b_self, b_nei, s_sc, ns);
    attn_kernel<<<NN / 16, 256, 0, stream>>>(s_sc, ns, t, coords, col, beta_mix,
                                             wred_bf, b_red, out);
}
